// Attention_15470472200716
// MI455X (gfx1250) — matrix operand order repaired and checked by reading, not compiled, not run on hardware
//
#include <hip/hip_runtime.h>
#include <hip/hip_bf16.h>

// ---------- types ----------
typedef _Float16 vh16 __attribute__((ext_vector_type(16)));
typedef _Float16 vh8  __attribute__((ext_vector_type(8)));
typedef float    vf8  __attribute__((ext_vector_type(8)));
typedef int      v4i_ __attribute__((vector_size(4 * sizeof(int))));

#define AS_GLOBAL __attribute__((address_space(1)))
#define AS_SHARED __attribute__((address_space(3)))

// dims (fixed by the reference problem)
#define BB   16
#define CIN  384
#define NN   1024
#define NH   8
#define KD   32
#define DD   128
#define DH   1024
#define COUT 384

// ---------- gfx1250 async global->LDS copy (guarded) ----------
#if defined(__has_builtin)
#if __has_builtin(__builtin_amdgcn_global_load_async_to_lds_b128) && \
    __has_builtin(__builtin_amdgcn_s_wait_asynccnt)
#define USE_ASYNC 1
#endif
#endif
#ifndef USE_ASYNC
#define USE_ASYNC 0
#endif

// copy 16 bytes global -> LDS (async when available)
__device__ __forceinline__ void copy16(const _Float16* g, _Float16* l) {
#if USE_ASYNC
    __builtin_amdgcn_global_load_async_to_lds_b128(
        (AS_GLOBAL v4i_*)(AS_GLOBAL void*)(g),
        (AS_SHARED v4i_*)(AS_SHARED void*)(l), 0, 0);
#else
    *(vh8*)l = *(const vh8*)g;
#endif
}
__device__ __forceinline__ void wait_async() {
#if USE_ASYNC
    __builtin_amdgcn_s_wait_asynccnt(0);
#endif
}

// ---------- WMMA fragment helpers (CDNA5 16x16x32 f16) ----------
// A (16x32, MxK): lane L: M = L%16; K chunks [hs*8, hs*8+8) and [16+hs*8, ...)
__device__ __forceinline__ vh16 fragA16x32(const _Float16* p, int ld) {
    unsigned lane = threadIdx.x & 31u;
    unsigned m = lane & 15u, hs = lane >> 4;
    const _Float16* q = p + (size_t)m * ld + hs * 8;
    union { vh16 v; vh8 h[2]; } r;
    r.h[0] = *(const vh8*)(q);
    r.h[1] = *(const vh8*)(q + 16);
    return r.v;
}
// B (32x16, KxN): lane L: column N = L%16; K chunks [hs*8, hs*8+8) and [16+hs*8, ...)
__device__ __forceinline__ vh16 fragB32x16(const _Float16* p, int ldcol) {
    unsigned lane = threadIdx.x & 31u;
    const _Float16* q = p + (size_t)(lane & 15u) * ldcol + (lane >> 4) * 8;
    union { vh16 v; vh8 h[2]; } r;
    r.h[0] = *(const vh8*)(q);
    r.h[1] = *(const vh8*)(q + 16);
    return r.v;
}
__device__ __forceinline__ vf8 wmma16(vh16 a, vh16 b, vf8 c) {
    return __builtin_amdgcn_wmma_f32_16x16x32_f16(false, a, false, b,
                                                  (short)0, c, false, false);
}

// =====================================================================
// Kernel 1: fused QKV projection + BN fold, fp32 -> fp16 outputs
// Y[1536,1024] = Wall[1536,384] @ X[b][384,1024]
// grid: (nTiles=8, oTiles=12, B=16), block 256 (8 waves), tile 128x128, K=32
// Software-pipelined: global loads for tile kc+1 overlap WMMAs of tile kc.
// =====================================================================
__global__ __launch_bounds__(256) void qkv_proj_kernel(
    const float* __restrict__ x,
    const float* __restrict__ wq, const float* __restrict__ bnq,
    const float* __restrict__ wk, const float* __restrict__ bnk,
    const float* __restrict__ wv, const float* __restrict__ bnv,
    _Float16* __restrict__ qb, _Float16* __restrict__ kb,
    _Float16* __restrict__ vb)
{
    __shared__ __align__(16) _Float16 As[128 * 32]; // [o][c]
    __shared__ __align__(16) _Float16 Bs[128 * 32]; // [n][c] (transposed)

    const int t  = threadIdx.x;
    const int bx = blockIdx.x;      // n tile
    const int by = blockIdx.y;      // o tile (0..11): 0-1 q, 2-3 k, 4-11 v
    const int bz = blockIdx.z;      // batch

    const float* W; const float* bn; int obase, mode, bnO;
    if (by < 2)      { W = wq; bn = bnq; obase = by * 128;        mode = 0; bnO = 256; }
    else if (by < 4) { W = wk; bn = bnk; obase = (by - 2) * 128;  mode = 1; bnO = 256; }
    else             { W = wv; bn = bnv; obase = (by - 4) * 128;  mode = 2; bnO = 1024; }

    const int w    = t >> 5;
    const int lane = t & 31;
    const int m    = lane & 15;
    const int hs   = lane >> 4;

    vf8 acc[8];
    for (int i = 0; i < 8; ++i)
        for (int j = 0; j < 8; ++j) acc[i][j] = 0.0f;

    const int arow = t >> 1, acoff = (t & 1) * 16;     // A loader mapping
    const int bcrow = t >> 3, bnoff = (t & 7) * 16;    // B loader mapping
    const float* srcA0 = W + (size_t)(obase + arow) * CIN + acoff;
    const float* srcB0 = x + ((size_t)bz * CIN + bcrow) * NN + bx * 128 + bnoff;

    float fA[16], fB[16];
    // prologue: fetch tile kc=0 into registers
    #pragma unroll
    for (int q4 = 0; q4 < 4; ++q4) {
        float4 a4 = ((const float4*)(srcA0))[q4];
        float4 b4 = ((const float4*)(srcB0))[q4];
        fA[q4*4+0]=a4.x; fA[q4*4+1]=a4.y; fA[q4*4+2]=a4.z; fA[q4*4+3]=a4.w;
        fB[q4*4+0]=b4.x; fB[q4*4+1]=b4.y; fB[q4*4+2]=b4.z; fB[q4*4+3]=b4.w;
    }

    for (int kc = 0; kc < CIN / 32; ++kc) {
        __syncthreads();                    // previous tile fully consumed
        {   // commit registers -> LDS (fp32 -> fp16; B transposed)
            vh8 h0, h1;
            #pragma unroll
            for (int j = 0; j < 8; ++j) { h0[j] = (_Float16)fA[j]; h1[j] = (_Float16)fA[8+j]; }
            *(vh8*)(&As[arow * 32 + acoff])     = h0;
            *(vh8*)(&As[arow * 32 + acoff + 8]) = h1;
            #pragma unroll
            for (int j = 0; j < 16; ++j)
                Bs[(bnoff + j) * 32 + bcrow] = (_Float16)fB[j];
        }
        __syncthreads();
        // prefetch tile kc+1 into registers (overlaps WMMA compute below)
        if (kc + 1 < CIN / 32) {
            const float* sa = srcA0 + (kc + 1) * 32;
            const float* sb = srcB0 + (size_t)(kc + 1) * 32 * NN;
            __builtin_prefetch(sa + 32, 0, 1);          // tile kc+2 hint
            __builtin_prefetch(sb + 32 * NN, 0, 1);
            #pragma unroll
            for (int q4 = 0; q4 < 4; ++q4) {
                float4 a4 = ((const float4*)(sa))[q4];
                float4 b4 = ((const float4*)(sb))[q4];
                fA[q4*4+0]=a4.x; fA[q4*4+1]=a4.y; fA[q4*4+2]=a4.z; fA[q4*4+3]=a4.w;
                fB[q4*4+0]=b4.x; fB[q4*4+1]=b4.y; fB[q4*4+2]=b4.z; fB[q4*4+3]=b4.w;
            }
        }
        // ---- compute: wave w owns o-rows [w*16, w*16+16) x all 128 n
        vh16 af = fragA16x32(&As[w * 16 * 32], 32);
        vh16 bf[8];
        #pragma unroll
        for (int tt = 0; tt < 8; ++tt) bf[tt] = fragB32x16(&Bs[tt * 16 * 32], 32);
        #pragma unroll
        for (int tt = 0; tt < 8; ++tt) acc[tt] = wmma16(af, bf[tt], acc[tt]);
    }

    // ---- epilogue: BN fold + scatter fp16 into attention layouts
    for (int i = 0; i < 8; ++i) {
        const int o_local = obase + w * 16 + 8 * hs + i;   // channel within this W
        const float g  = bn[0 * bnO + o_local];
        const float b_ = bn[1 * bnO + o_local];
        const float mu = bn[2 * bnO + o_local];
        const float vv = bn[3 * bnO + o_local];
        const float s    = g * rsqrtf(vv + 1e-5f);
        const float bias = b_ - mu * s;
        for (int tt = 0; tt < 8; ++tt) {
            const int n = bx * 128 + tt * 16 + m;
            const float y = acc[tt][i] * s + bias;
            if (mode == 2) {
                vb[((size_t)bz * DH + o_local) * NN + n] = (_Float16)y;
            } else {
                const size_t idx = (((size_t)bz * NH + (o_local >> 5)) * NN + n) * KD
                                   + (o_local & 31);
                if (mode == 0) qb[idx] = (_Float16)y;
                else           kb[idx] = (_Float16)y;
            }
        }
    }
}

// =====================================================================
// Kernel 2: flash attention. grid (N/128=8, B*nh=128), block 256 (8 waves).
// k/v tiles staged once per block into double-buffered LDS with async
// global->LDS copies; copy of tile j+1 overlaps WMMAs of tile j.
// Each wave: 16 query rows, online softmax, O[16,128] f32 in registers.
// =====================================================================
__global__ __launch_bounds__(256) void flash_attn_kernel(
    const _Float16* __restrict__ qb, const _Float16* __restrict__ kb,
    const _Float16* __restrict__ vb, _Float16* __restrict__ xxb)
{
    __shared__ __align__(16) _Float16 KsS[2][32 * 32];   // [n_local][kd]
    __shared__ __align__(16) _Float16 VsS[2][128 * 32];  // [d][n_local]
    __shared__ __align__(16) _Float16 Ps[8][16 * 32];    // per-wave P staging

    const int t = threadIdx.x;
    const int w = t >> 5, lane = t & 31, m = lane & 15, hs = lane >> 4;
    const int bh = blockIdx.y;              // b*8 + h
    const int b  = bh >> 3, h = bh & 7;
    const int q0 = blockIdx.x * 128 + w * 16;

    const _Float16* kbh   = kb + (size_t)bh * NN * KD;            // [n][kd]
    const _Float16* vhead = vb + ((size_t)b * DH + h * DD) * NN;  // [d][N]

    // q A-fragment: rows q0..q0+15, K = kd = 32 (one WMMA K-step), loaded once
    const vh16 qf = fragA16x32(qb + ((size_t)bh * NN + q0) * KD, KD);

    // sqrt(kd) * log2(e): reference divides by scale=kd^-0.5, softmax via exp2
    const float ALPHA = 5.65685424949238f * 1.44269504088896f;

    float mrow[8], lrow[8];
    vf8 o[8];
    for (int i = 0; i < 8; ++i) { mrow[i] = -1e30f; lrow[i] = 0.0f; }
    for (int tt = 0; tt < 8; ++tt)
        for (int i = 0; i < 8; ++i) o[tt][i] = 0.0f;

    _Float16* myP = &Ps[w][0];

    // cooperative staging of one 32-column k/v tile into LDS buffer `buf`
    auto stage = [&](int buf, int j2) {
        const int n0 = j2 * 32;
        if (t < 128)   // k tile: 32 rows x 32 halves, fully contiguous (2 KB)
            copy16(kbh + (size_t)n0 * KD + t * 8, &KsS[buf][t * 8]);
        #pragma unroll
        for (int c = t; c < 512; c += 256) {   // v tile: 128 rows x 64 B (8 KB)
            const int d = c >> 2, part = c & 3;
            copy16(vhead + (size_t)d * NN + n0 + part * 8,
                   &VsS[buf][d * 32 + part * 8]);
        }
    };

    stage(0, 0);   // prologue

    for (int j2 = 0; j2 < NN / 32; ++j2) {
        wait_async();        // this wave's copies for tile j2 have landed
        __syncthreads();     // all waves' copies landed; prev tile consumed
        if (j2 + 1 < NN / 32) stage((j2 + 1) & 1, j2 + 1);  // overlap next copy

        const _Float16* Kc = &KsS[j2 & 1][0];
        const _Float16* Vc = &VsS[j2 & 1][0];

        // S = q @ k^T for 32 key columns (two 16-col WMMA tiles)
        vh16 kf0 = fragB32x16(Kc, KD);
        vh16 kf1 = fragB32x16(Kc + 16 * KD, KD);
        vf8 s0, s1;
        for (int i = 0; i < 8; ++i) { s0[i] = 0.0f; s1[i] = 0.0f; }
        s0 = wmma16(qf, kf0, s0);
        s1 = wmma16(qf, kf1, s1);

        // scale into log2 domain + online softmax stats
        float resc[8];
        for (int i = 0; i < 8; ++i) {
            float a = s0[i] * ALPHA, c = s1[i] * ALPHA;
            s0[i] = a; s1[i] = c;
            float r = fmaxf(a, c);
            r = fmaxf(r, __shfl_xor(r, 1, 16));
            r = fmaxf(r, __shfl_xor(r, 2, 16));
            r = fmaxf(r, __shfl_xor(r, 4, 16));
            r = fmaxf(r, __shfl_xor(r, 8, 16));
            const float mn = fmaxf(mrow[i], r);
            resc[i] = exp2f(mrow[i] - mn);
            mrow[i] = mn;
        }
        // P = exp2(S - m), row sums, stage P (fp16) to LDS for A-layout reload
        for (int i = 0; i < 8; ++i) {
            const float p0 = exp2f(s0[i] - mrow[i]);
            const float p1 = exp2f(s1[i] - mrow[i]);
            float rs = p0 + p1;
            rs += __shfl_xor(rs, 1, 16);
            rs += __shfl_xor(rs, 2, 16);
            rs += __shfl_xor(rs, 4, 16);
            rs += __shfl_xor(rs, 8, 16);
            lrow[i] = lrow[i] * resc[i] + rs;
            const int r = i + 8 * hs;                 // C-layout row of this lane
            myP[r * 32 + m]      = (_Float16)p0;
            myP[r * 32 + 16 + m] = (_Float16)p1;
        }
        // rescale O accumulators
        for (int tt = 0; tt < 8; ++tt)
            for (int i = 0; i < 8; ++i) o[tt][i] *= resc[i];

        // P as A-fragment (same-wave LDS store->load is in-order)
        vh16 pf = fragA16x32(myP, 32);
        // O += P @ V  (8 d-tiles of 16)
        #pragma unroll
        for (int tt = 0; tt < 8; ++tt) {
            vh16 vf = fragB32x16(Vc + tt * 16 * 32, 32);
            o[tt] = wmma16(pf, vf, o[tt]);
        }
    }

    // normalize, store xx as fp16 [B, dh, N] (d-major for the output GEMM)
    for (int i = 0; i < 8; ++i) {
        const float inv = 1.0f / lrow[i];
        const int qn = q0 + i + 8 * hs;
        for (int tt = 0; tt < 8; ++tt) {
            const int d = h * DD + tt * 16 + m;
            xxb[((size_t)b * DH + d) * NN + qn] = (_Float16)(o[tt][i] * inv);
        }
    }
}

// =====================================================================
// Kernel 3: output projection. out[b][384,1024] = wp[384,1024] @ xx[b][1024,1024]
// grid (nTiles=8, oTiles=3, B=16), block 256, tile 128x128, K-step 32.
// Software-pipelined like kernel 1.
// =====================================================================
__global__ __launch_bounds__(256) void out_proj_kernel(
    const _Float16* __restrict__ xxb, const float* __restrict__ wp,
    const float* __restrict__ bnp, float* __restrict__ out)
{
    __shared__ __align__(16) _Float16 As[128 * 32]; // [o][c]
    __shared__ __align__(16) _Float16 Bs[128 * 32]; // [n][c]

    const int t  = threadIdx.x;
    const int bx = blockIdx.x, by = blockIdx.y, bz = blockIdx.z;
    const int w = t >> 5, lane = t & 31, m = lane & 15, hs = lane >> 4;

    vf8 acc[8];
    for (int i = 0; i < 8; ++i)
        for (int j = 0; j < 8; ++j) acc[i][j] = 0.0f;

    const int arow = t >> 1, acoff = (t & 1) * 16;
    const int bcrow = t >> 3, bnoff = (t & 7) * 16;
    const float*    srcA0 = wp + (size_t)(by * 128 + arow) * DH + acoff;
    const _Float16* srcB0 = xxb + ((size_t)bz * DH + bcrow) * NN + bx * 128 + bnoff;

    float fA[16];
    vh8 u0, u1;
    #pragma unroll
    for (int q4 = 0; q4 < 4; ++q4) {
        float4 a4 = ((const float4*)(srcA0))[q4];
        fA[q4*4+0]=a4.x; fA[q4*4+1]=a4.y; fA[q4*4+2]=a4.z; fA[q4*4+3]=a4.w;
    }
    u0 = ((const vh8*)srcB0)[0];
    u1 = ((const vh8*)srcB0)[1];

    for (int kc = 0; kc < DH / 32; ++kc) {
        __syncthreads();
        {   // commit registers -> LDS
            vh8 h0, h1;
            #pragma unroll
            for (int j = 0; j < 8; ++j) { h0[j] = (_Float16)fA[j]; h1[j] = (_Float16)fA[8+j]; }
            *(vh8*)(&As[arow * 32 + acoff])     = h0;
            *(vh8*)(&As[arow * 32 + acoff + 8]) = h1;
            #pragma unroll
            for (int j = 0; j < 8; ++j) {
                Bs[(bnoff + j) * 32 + bcrow]     = u0[j];
                Bs[(bnoff + 8 + j) * 32 + bcrow] = u1[j];
            }
        }
        __syncthreads();
        if (kc + 1 < DH / 32) {     // prefetch next tile into registers
            const float*    sa = srcA0 + (kc + 1) * 32;
            const _Float16* sb = srcB0 + (size_t)(kc + 1) * 32 * NN;
            __builtin_prefetch(sa + 32, 0, 1);
            __builtin_prefetch(sb + 32 * NN, 0, 1);
            #pragma unroll
            for (int q4 = 0; q4 < 4; ++q4) {
                float4 a4 = ((const float4*)(sa))[q4];
                fA[q4*4+0]=a4.x; fA[q4*4+1]=a4.y; fA[q4*4+2]=a4.z; fA[q4*4+3]=a4.w;
            }
            u0 = ((const vh8*)sb)[0];
            u1 = ((const vh8*)sb)[1];
        }
        vh16 af = fragA16x32(&As[w * 16 * 32], 32);
        vh16 bf[8];
        #pragma unroll
        for (int tt = 0; tt < 8; ++tt) bf[tt] = fragB32x16(&Bs[tt * 16 * 32], 32);
        #pragma unroll
        for (int tt = 0; tt < 8; ++tt) acc[tt] = wmma16(af, bf[tt], acc[tt]);
    }

    for (int i = 0; i < 8; ++i) {
        const int o = by * 128 + w * 16 + 8 * hs + i;
        const float g  = bnp[0 * COUT + o];
        const float b_ = bnp[1 * COUT + o];
        const float mu = bnp[2 * COUT + o];
        const float vv = bnp[3 * COUT + o];
        const float s    = g * rsqrtf(vv + 1e-5f);
        const float bias = b_ - mu * s;
        for (int tt = 0; tt < 8; ++tt) {
            const int n = bx * 128 + tt * 16 + m;
            out[((size_t)bz * COUT + o) * NN + n] = acc[tt][i] * s + bias;
        }
    }
}

// =====================================================================
extern "C" void kernel_launch(void* const* d_in, const int* in_sizes, int n_in,
                              void* d_out, int out_size, void* d_ws, size_t ws_size,
                              hipStream_t stream) {
    (void)in_sizes; (void)n_in; (void)out_size; (void)ws_size;
    const float* x   = (const float*)d_in[0];
    const float* wq  = (const float*)d_in[1];
    const float* bnq = (const float*)d_in[2];
    const float* wk  = (const float*)d_in[3];
    const float* bnk = (const float*)d_in[4];
    const float* wv  = (const float*)d_in[5];
    const float* bnv = (const float*)d_in[6];
    const float* wp  = (const float*)d_in[7];
    const float* bnp = (const float*)d_in[8];

    char* ws = (char*)d_ws;
    _Float16* qb  = (_Float16*)(ws);                              //  8 MB
    _Float16* kb  = (_Float16*)(ws + (size_t)8  * 1024 * 1024);   //  8 MB
    _Float16* vb  = (_Float16*)(ws + (size_t)16 * 1024 * 1024);   // 32 MB
    _Float16* xxb = (_Float16*)(ws + (size_t)48 * 1024 * 1024);   // 32 MB
    float* out = (float*)d_out;

    qkv_proj_kernel<<<dim3(8, 12, 16), 256, 0, stream>>>(
        x, wq, bnq, wk, bnk, wv, bnv, qb, kb, vb);
    flash_attn_kernel<<<dim3(8, 128, 1), 256, 0, stream>>>(qb, kb, vb, xxb);
    out_proj_kernel<<<dim3(8, 3, 16), 256, 0, stream>>>(xxb, wp, bnp, out);
}
